// GraspDenoiser_36129264894466
// MI455X (gfx1250) — matrix operand order repaired and checked by reading, not compiled, not run on hardware
//
#include <hip/hip_runtime.h>

// ---------------------------------------------------------------------------
// GraspDenoiser forward for MI455X (gfx1250): bf16 WMMA GEMM path, v2.
//  - FiLM + jid-scatter fused into GEMM epilogues (saves ~0.9 GB HBM traffic)
//  - 32x16 per-wave tile: two independent accumulators break the WMMA RAW
//    chain and halve B-fragment traffic.
// ---------------------------------------------------------------------------

typedef __attribute__((ext_vector_type(16))) __bf16 v16b;
typedef __attribute__((ext_vector_type(8)))  __bf16 v8b;
typedef __attribute__((ext_vector_type(8)))  float  v8f;

static constexpr int Bn = 4096;
static constexpr int Tt = 23;
static constexpr int Jj = 22;
static constexpr int Dd = 128;
static constexpr int Hh = 8;
static constexpr int Ll = 4;
static constexpr int NBb = 3;
static constexpr int MT = Bn * Tt;   // 94208  (multiple of 32)
static constexpr int MJ = Bn * Jj;   // 90112  (multiple of 32)

__device__ __forceinline__ float gelu_tanh(float x) {
    const float k = 0.7978845608028654f;
    float x3 = x * x * x;
    return 0.5f * x * (1.0f + tanhf(k * (x + 0.044715f * x3)));
}

// ---------------------------------------------------------------------------
// Generic bf16 WMMA GEMM:  out = act(A[M,K] * WT[N,K]^T + bias)
// grid = (M/32, N/128), block = 256 (8 waves). Each wave: 32(M) x 16(N) tile,
// two accumulators (independent WMMA chains) sharing one B fragment.
// Epilogue modes:
//   film != null : outF[idx] = res[idx] + v*(1+gamma) + beta   (gamma/beta from
//                  film[(r/Tt)*256 + {c, 128+c}])
//   jid  != null : embedder scatter: v += jid[(r%Jj)*128+c];
//                  outF[((r/Jj)*Tt + 1 + r%Jj)*128 + c] = v
//   else         : optional res add, fp32 and/or bf16 store.
// act: 0 = none, 1 = gelu, 2 = relu
// ---------------------------------------------------------------------------
__global__ void gemm_bf16_wmma(const __bf16* __restrict__ A,
                               const __bf16* __restrict__ WT,
                               const float* __restrict__ bias,
                               float* __restrict__ outF,
                               __bf16* __restrict__ outB,
                               const float* __restrict__ res,
                               const float* __restrict__ film,
                               const float* __restrict__ jid,
                               int M, int N, int K, int act) {
    const int lane = threadIdx.x & 31;
    const int wave = threadIdx.x >> 5;
    const int row0 = blockIdx.x * 32;
    const int col0 = blockIdx.y * 128 + wave * 16;
    if (col0 >= N) return;

    const int m  = lane & 15;
    const int hh = lane >> 4;

    const __bf16* arow0 = A  + (size_t)(row0 + m) * K;
    const __bf16* arow1 = A  + (size_t)(row0 + 16 + m) * K;
    const __bf16* brow  = WT + (size_t)(col0 + m) * K;

    v8f acc0 = {}, acc1 = {};
    for (int k0 = 0; k0 < K; k0 += 32) {
        // A fragment: 16x32, lane m = M-row; hh=0 -> K {0..7,16..23}, hh=1 -> {8..15,24..31}
        v8b a0lo = *(const v8b*)(arow0 + k0 + hh * 8);
        v8b a0hi = *(const v8b*)(arow0 + k0 + 16 + hh * 8);
        v8b a1lo = *(const v8b*)(arow1 + k0 + hh * 8);
        v8b a1hi = *(const v8b*)(arow1 + k0 + 16 + hh * 8);
        // B fragment: 32x16, lane m = N-col; hh=0 -> K {0..7,16..23}, hh=1 -> {8..15,24..31}
        v8b blo  = *(const v8b*)(brow + k0 + hh * 8);
        v8b bhi  = *(const v8b*)(brow + k0 + 16 + hh * 8);
        v16b av0, av1, bv;
#pragma unroll
        for (int i = 0; i < 8; ++i) {
            av0[i] = a0lo[i]; av0[8 + i] = a0hi[i];
            av1[i] = a1lo[i]; av1[8 + i] = a1hi[i];
            bv[i]  = blo[i];  bv[8 + i]  = bhi[i];
        }
        acc0 = __builtin_amdgcn_wmma_f32_16x16x32_bf16(
            false, av0, false, bv, (short)0, acc0, false, false);
        acc1 = __builtin_amdgcn_wmma_f32_16x16x32_bf16(
            false, av1, false, bv, (short)0, acc1, false, false);
    }

#pragma unroll
    for (int t = 0; t < 16; ++t) {
        int e = t & 7;
        int r = row0 + e + hh * 8 + (t >> 3) * 16;  // C/D: lanes 0-15 M=e, 16-31 M=e+8
        int c = col0 + m;
        float v = (t < 8) ? acc0[e] : acc1[e];
        if (bias) v += bias[c];
        if (act == 1) v = gelu_tanh(v);
        else if (act == 2) v = fmaxf(v, 0.0f);
        size_t idx = (size_t)r * N + c;
        if (film) {
            int bb = r / Tt;
            float gamma = film[(size_t)bb * 256 + c];
            float beta  = film[(size_t)bb * 256 + 128 + c];
            outF[idx] = res[idx] + v * (1.0f + gamma) + beta;
        } else if (jid) {
            int bb = r / Jj, jj = r % Jj;
            v += jid[jj * 128 + c];
            outF[((size_t)(bb * Tt) + 1 + jj) * 128 + c] = v;
        } else {
            if (res)  v += res[idx];
            if (outF) outF[idx] = v;
            if (outB) outB[idx] = (__bf16)v;
        }
    }
}

// ---------------------------------------------------------------------------
// LayerNorm over D=128, one wave32 per row, bf16 output.
// ---------------------------------------------------------------------------
__global__ void ln_to_bf16(const float* __restrict__ x,
                           const float* __restrict__ g,
                           const float* __restrict__ b,
                           __bf16* __restrict__ y, int Mrows) {
    int row = blockIdx.x * 8 + (threadIdx.x >> 5);
    if (row >= Mrows) return;
    int lane = threadIdx.x & 31;
    const float4 v = *(const float4*)(x + (size_t)row * 128 + lane * 4);
    float s = v.x + v.y + v.z + v.w;
#pragma unroll
    for (int o = 16; o; o >>= 1) s += __shfl_xor(s, o, 32);
    float mean = s * (1.0f / 128.0f);
    float dx = v.x - mean, dy = v.y - mean, dz = v.z - mean, dw = v.w - mean;
    float q = dx * dx + dy * dy + dz * dz + dw * dw;
#pragma unroll
    for (int o = 16; o; o >>= 1) q += __shfl_xor(q, o, 32);
    float rstd = rsqrtf(q * (1.0f / 128.0f) + 1e-5f);
    int c = lane * 4;
    __bf16* yr = y + (size_t)row * 128 + c;
    yr[0] = (__bf16)(dx * rstd * g[c + 0] + b[c + 0]);
    yr[1] = (__bf16)(dy * rstd * g[c + 1] + b[c + 1]);
    yr[2] = (__bf16)(dz * rstd * g[c + 2] + b[c + 2]);
    yr[3] = (__bf16)(dw * rstd * g[c + 3] + b[c + 3]);
}

// ---------------------------------------------------------------------------
// Attention: one thread per (b, head, i). T=23, DH=16.
// qkv is bf16 (MT, 384) = [q|k|v]; out o bf16 (MT, 128).
// ---------------------------------------------------------------------------
__global__ void attn_kernel(const __bf16* __restrict__ qkv,
                            const float* __restrict__ dist,
                            const float* __restrict__ bscale,
                            __bf16* __restrict__ o) {
    int idx = blockIdx.x * blockDim.x + threadIdx.x;
    if (idx >= Bn * Hh * Tt) return;
    int i  = idx % Tt;
    int hd = (idx / Tt) % Hh;
    int b  = idx / (Tt * Hh);

    const __bf16* qp = qkv + (size_t)(b * Tt + i) * 384 + hd * 16;
    float q[16];
#pragma unroll
    for (int d = 0; d < 16; ++d) q[d] = (float)qp[d];

    float bs = bscale[hd];
    float sc = (bs > 20.0f) ? bs : log1pf(expf(bs));   // softplus

    const float* drow = dist + ((size_t)b * Tt + i) * Tt;
    float scores[Tt];
    float mx = -1e30f;
    for (int j = 0; j < Tt; ++j) {
        const __bf16* kp = qkv + (size_t)(b * Tt + j) * 384 + 128 + hd * 16;
        float s = 0.0f;
#pragma unroll
        for (int d = 0; d < 16; ++d) s += q[d] * (float)kp[d];
        s = s * 0.25f - sc * drow[j];                  // 1/sqrt(16) = 0.25
        scores[j] = s;
        mx = fmaxf(mx, s);
    }
    float denom = 0.0f;
    for (int j = 0; j < Tt; ++j) { scores[j] = expf(scores[j] - mx); denom += scores[j]; }
    float inv = 1.0f / denom;

    float outv[16];
#pragma unroll
    for (int d = 0; d < 16; ++d) outv[d] = 0.0f;
    for (int j = 0; j < Tt; ++j) {
        float p = scores[j] * inv;
        const __bf16* vp = qkv + (size_t)(b * Tt + j) * 384 + 256 + hd * 16;
#pragma unroll
        for (int d = 0; d < 16; ++d) outv[d] += p * (float)vp[d];
    }
    __bf16* op = o + (size_t)(b * Tt + i) * 128 + hd * 16;
#pragma unroll
    for (int d = 0; d < 16; ++d) op[d] = (__bf16)outv[d];
}

// ---------------------------------------------------------------------------
// Small helper kernels
// ---------------------------------------------------------------------------
__global__ void wconv(const float* __restrict__ src, __bf16* __restrict__ dst,
                      int K, int N) {           // src K x N -> dst N x K (bf16)
    int idx = blockIdx.x * blockDim.x + threadIdx.x;
    if (idx >= K * N) return;
    int k = idx / N, n = idx % N;
    dst[(size_t)n * K + k] = (__bf16)src[idx];
}

__global__ void cvt_bf16(const float* __restrict__ x, __bf16* __restrict__ y, int n) {
    int idx = blockIdx.x * blockDim.x + threadIdx.x;
    if (idx < n) y[idx] = (__bf16)x[idx];
}

__global__ void dist_kernel(const float* __restrict__ anchor, float* __restrict__ dist) {
    int idx = blockIdx.x * blockDim.x + threadIdx.x;
    if (idx >= Bn * Tt * Tt) return;
    int j = idx % Tt, i = (idx / Tt) % Tt, b = idx / (Tt * Tt);
    const float* pa = anchor + ((size_t)b * Tt + i) * 3;
    const float* pb = anchor + ((size_t)b * Tt + j) * 3;
    float dx = pa[0] - pb[0], dy = pa[1] - pb[1], dz = pa[2] - pb[2];
    dist[idx] = sqrtf(dx * dx + dy * dy + dz * dz + 1e-8f);
}

__global__ void wrist_kernel(const float* __restrict__ xyz, const float* __restrict__ rot6d,
                             const float* __restrict__ W, const float* __restrict__ bias,
                             float* __restrict__ h) {
    int idx = blockIdx.x * blockDim.x + threadIdx.x;
    if (idx >= Bn * 128) return;
    int d = idx & 127, b = idx >> 7;
    float in9[9];
    in9[0] = xyz[b * 3 + 0]; in9[1] = xyz[b * 3 + 1]; in9[2] = xyz[b * 3 + 2];
#pragma unroll
    for (int k = 0; k < 6; ++k) in9[3 + k] = rot6d[b * 6 + k];
    float s = bias[d];
#pragma unroll
    for (int k = 0; k < 9; ++k) s += in9[k] * W[k * 128 + d];
    h[(size_t)(b * Tt) * 128 + d] = s;
}

__global__ void ang_embed(const float* __restrict__ angles, const float* __restrict__ W1,
                          const float* __restrict__ b1, __bf16* __restrict__ a1) {
    int idx = blockIdx.x * blockDim.x + threadIdx.x;
    if (idx >= MJ * 128) return;
    int d = idx & 127, bj = idx >> 7;
    float v = angles[bj] * W1[d] + b1[d];
    a1[idx] = (__bf16)fmaxf(v, 0.0f);
}

__global__ void proj_head(const __bf16* __restrict__ x, const float* __restrict__ W,
                          const float* __restrict__ bias, float* __restrict__ out,
                          int M, int C, int ld, int off) {
    int idx = blockIdx.x * blockDim.x + threadIdx.x;
    if (idx >= M * C) return;
    int c = idx % C, mrow = idx / C;
    const __bf16* xr = x + (size_t)mrow * 128;
    float s = bias[c];
#pragma unroll 16
    for (int d = 0; d < 128; ++d) s += (float)xr[d] * W[d * C + c];
    out[(size_t)mrow * ld + off + c] = s;
}

// ---------------------------------------------------------------------------
extern "C" void kernel_launch(void* const* d_in, const int* in_sizes, int n_in,
                              void* d_out, int out_size, void* d_ws, size_t ws_size,
                              hipStream_t stream) {
    (void)in_sizes; (void)n_in; (void)out_size; (void)ws_size;
    auto F = [&](int i) { return (const float*)d_in[i]; };

    // inputs
    const float* xyz        = F(0);
    const float* rot6d      = F(1);
    const float* angles     = F(2);
    const float* xyz_anchor = F(3);
    const float* scene      = F(4);
    // params (setup_inputs dict order)
    const float* wrist_W = F(5),  *wrist_b = F(6);
    const float* ang1_W  = F(7),  *ang1_b  = F(8);
    const float* ang2_W  = F(9),  *ang2_b  = F(10);
    const float* jid     = F(11);
    const float* qkv_W   = F(12), *qkv_b   = F(13);
    const float* out_W   = F(14), *out_b   = F(15);
    const float* ln1_g   = F(16), *ln1_b   = F(17);
    const float* ln2_g   = F(18), *ln2_b   = F(19);
    const float* fc1_W   = F(20), *fc1_b   = F(21);
    const float* fc2_W   = F(22), *fc2_b   = F(23);
    const float* film_W  = F(24), *film_b  = F(25);
    const float* bias_scale = F(26);
    const float* pb_ln_g = F(27), *pb_ln_b = F(28);
    const float* pb_W1   = F(29), *pb_b1   = F(30);
    const float* pb_W2   = F(31), *pb_b2   = F(32);
    const float* ph_ln_g = F(33), *ph_ln_b = F(34);
    const float* pos_W   = F(35), *pos_b   = F(36);
    const float* rb_ln_g = F(37), *rb_ln_b = F(38);
    const float* rb_W1   = F(39), *rb_b1   = F(40);
    const float* rb_W2   = F(41), *rb_b2   = F(42);
    const float* rh_ln_g = F(43), *rh_ln_b = F(44);
    const float* rot_W   = F(45), *rot_b   = F(46);
    const float* ah_W1   = F(47), *ah_b1   = F(48);
    const float* ah_W2   = F(49), *ah_b2   = F(50);

    float* out = (float*)d_out;

    // workspace carve-out
    char* p = (char*)d_ws;
    auto alloc = [&](size_t bytes) { char* r = p; p += (bytes + 255) & ~(size_t)255; return r; };
    float*  h     = (float*)alloc((size_t)MT * 128 * 4);
    float*  hp    = (float*)alloc((size_t)MT * 128 * 4);
    float*  hr    = (float*)alloc((size_t)MT * 128 * 4);
    float*  gb    = (float*)alloc((size_t)Bn * 256 * 4);
    float*  dist  = (float*)alloc((size_t)Bn * Tt * Tt * 4);
    __bf16* xbf   = (__bf16*)alloc((size_t)MT * 128 * 2);
    __bf16* ybf   = (__bf16*)alloc((size_t)MT * 512 * 2);
    __bf16* obf   = (__bf16*)alloc((size_t)MT * 128 * 2);
    __bf16* qkvbf = (__bf16*)alloc((size_t)MT * 384 * 2);
    __bf16* a1bf  = (__bf16*)alloc((size_t)MJ * 128 * 2);
    __bf16* scbf  = (__bf16*)alloc((size_t)Bn * 128 * 2);
    // transposed bf16 weights
    __bf16* qkvT  = (__bf16*)alloc((size_t)Ll * 49152 * 2);
    __bf16* outT  = (__bf16*)alloc((size_t)Ll * 16384 * 2);
    __bf16* fc1T  = (__bf16*)alloc((size_t)Ll * 32768 * 2);
    __bf16* fc2T  = (__bf16*)alloc((size_t)Ll * 32768 * 2);
    __bf16* filmT = (__bf16*)alloc((size_t)Ll * 32768 * 2);
    __bf16* ang2T = (__bf16*)alloc((size_t)16384 * 2);
    __bf16* ahW1T = (__bf16*)alloc((size_t)16384 * 2);
    __bf16* pbW1T = (__bf16*)alloc((size_t)NBb * 65536 * 2);
    __bf16* pbW2T = (__bf16*)alloc((size_t)NBb * 65536 * 2);
    __bf16* rbW1T = (__bf16*)alloc((size_t)NBb * 65536 * 2);
    __bf16* rbW2T = (__bf16*)alloc((size_t)NBb * 65536 * 2);

    auto conv = [&](const float* src, __bf16* dst, int K, int N) {
        wconv<<<(K * N + 255) / 256, 256, 0, stream>>>(src, dst, K, N);
    };
    auto gemm = [&](const __bf16* A, const __bf16* WT, const float* bias,
                    float* oF, __bf16* oB, const float* res,
                    const float* film, const float* jidp,
                    int M, int N, int K, int act) {
        gemm_bf16_wmma<<<dim3(M / 32, N / 128), 256, 0, stream>>>(
            A, WT, bias, oF, oB, res, film, jidp, M, N, K, act);
    };

    // --- weight prep: fp32 KxN -> bf16 NxK
    for (int l = 0; l < Ll; ++l) {
        conv(qkv_W  + (size_t)l * 49152, qkvT  + (size_t)l * 49152, 128, 384);
        conv(out_W  + (size_t)l * 16384, outT  + (size_t)l * 16384, 128, 128);
        conv(fc1_W  + (size_t)l * 32768, fc1T  + (size_t)l * 32768, 128, 256);
        conv(fc2_W  + (size_t)l * 32768, fc2T  + (size_t)l * 32768, 256, 128);
        conv(film_W + (size_t)l * 32768, filmT + (size_t)l * 32768, 128, 256);
    }
    conv(ang2_W, ang2T, 128, 128);
    conv(ah_W1, ahW1T, 128, 128);
    for (int i = 0; i < NBb; ++i) {
        conv(pb_W1 + (size_t)i * 65536, pbW1T + (size_t)i * 65536, 128, 512);
        conv(pb_W2 + (size_t)i * 65536, pbW2T + (size_t)i * 65536, 512, 128);
        conv(rb_W1 + (size_t)i * 65536, rbW1T + (size_t)i * 65536, 128, 512);
        conv(rb_W2 + (size_t)i * 65536, rbW2T + (size_t)i * 65536, 512, 128);
    }

    // --- embedder -> h (B, T, 128); jid-add + token scatter fused in epilogue
    ang_embed<<<(MJ * 128 + 255) / 256, 256, 0, stream>>>(angles, ang1_W, ang1_b, a1bf);
    gemm(a1bf, ang2T, ang2_b, h, nullptr, nullptr, nullptr, jid, MJ, 128, 128, 0);
    wrist_kernel<<<(Bn * 128 + 255) / 256, 256, 0, stream>>>(xyz, rot6d, wrist_W, wrist_b, h);

    // --- distance bias + scene bf16
    dist_kernel<<<(Bn * Tt * Tt + 255) / 256, 256, 0, stream>>>(xyz_anchor, dist);
    cvt_bf16<<<(Bn * 128 + 255) / 256, 256, 0, stream>>>(scene, scbf, Bn * 128);

    // --- transformer layers
    for (int l = 0; l < Ll; ++l) {
        ln_to_bf16<<<MT / 8, 256, 0, stream>>>(h, ln1_g + l * 128, ln1_b + l * 128, xbf, MT);
        gemm(xbf, qkvT + (size_t)l * 49152, qkv_b + l * 384, nullptr, qkvbf, nullptr,
             nullptr, nullptr, MT, 384, 128, 0);
        attn_kernel<<<(Bn * Hh * Tt + 255) / 256, 256, 0, stream>>>(
            qkvbf, dist, bias_scale + l * Hh, obf);
        gemm(obf, outT + (size_t)l * 16384, out_b + l * 128, h, nullptr, h,
             nullptr, nullptr, MT, 128, 128, 0);
        gemm(scbf, filmT + (size_t)l * 32768, film_b + l * 256, gb, nullptr, nullptr,
             nullptr, nullptr, Bn, 256, 128, 0);
        ln_to_bf16<<<MT / 8, 256, 0, stream>>>(h, ln2_g + l * 128, ln2_b + l * 128, xbf, MT);
        gemm(xbf, fc1T + (size_t)l * 32768, fc1_b + l * 256, nullptr, ybf, nullptr,
             nullptr, nullptr, MT, 256, 128, 1);
        // fc2 with fused FiLM residual: h = h + fc2(y)*(1+gamma) + beta
        gemm(ybf, fc2T + (size_t)l * 32768, fc2_b + l * 128, h, nullptr, h,
             gb, nullptr, MT, 128, 256, 0);
    }

    // --- heads
    hipMemcpyAsync(hp, h, (size_t)MT * 128 * 4, hipMemcpyDeviceToDevice, stream);
    hipMemcpyAsync(hr, h, (size_t)MT * 128 * 4, hipMemcpyDeviceToDevice, stream);

    for (int i = 0; i < NBb; ++i) {
        ln_to_bf16<<<MT / 8, 256, 0, stream>>>(hp, pb_ln_g + i * 128, pb_ln_b + i * 128, xbf, MT);
        gemm(xbf, pbW1T + (size_t)i * 65536, pb_b1 + i * 512, nullptr, ybf, nullptr,
             nullptr, nullptr, MT, 512, 128, 1);
        gemm(ybf, pbW2T + (size_t)i * 65536, pb_b2 + i * 128, hp, nullptr, hp,
             nullptr, nullptr, MT, 128, 512, 0);
    }
    ln_to_bf16<<<MT / 8, 256, 0, stream>>>(hp, ph_ln_g, ph_ln_b, xbf, MT);
    proj_head<<<(MT * 3 + 255) / 256, 256, 0, stream>>>(xbf, pos_W, pos_b, out, MT, 3, 10, 0);

    for (int i = 0; i < NBb; ++i) {
        ln_to_bf16<<<MT / 8, 256, 0, stream>>>(hr, rb_ln_g + i * 128, rb_ln_b + i * 128, xbf, MT);
        gemm(xbf, rbW1T + (size_t)i * 65536, rb_b1 + i * 512, nullptr, ybf, nullptr,
             nullptr, nullptr, MT, 512, 128, 1);
        gemm(ybf, rbW2T + (size_t)i * 65536, rb_b2 + i * 128, hr, nullptr, hr,
             nullptr, nullptr, MT, 128, 512, 0);
    }
    ln_to_bf16<<<MT / 8, 256, 0, stream>>>(hr, rh_ln_g, rh_ln_b, xbf, MT);
    proj_head<<<(MT * 6 + 255) / 256, 256, 0, stream>>>(xbf, rot_W, rot_b, out, MT, 6, 10, 3);

    // angle head: relu(h @ ah_W1 + b1) @ ah_W2 + b2
    cvt_bf16<<<(MT * 128 + 255) / 256, 256, 0, stream>>>(h, xbf, MT * 128);
    gemm(xbf, ahW1T, ah_b1, nullptr, obf, nullptr, nullptr, nullptr, MT, 128, 128, 2);
    proj_head<<<(MT * 1 + 255) / 256, 256, 0, stream>>>(obf, ah_W2, ah_b2, out, MT, 1, 10, 9);
}
